// MultiHeadedAttentionKVCache_37117107372630
// MI455X (gfx1250) — matrix operand order repaired and checked by reading, not compiled, not run on hardware
//
#include <hip/hip_runtime.h>

typedef __attribute__((ext_vector_type(16))) _Float16 v16h;
typedef __attribute__((ext_vector_type(4)))  _Float16 v4h;
typedef __attribute__((ext_vector_type(8)))  float    v8f;
typedef __attribute__((ext_vector_type(4)))  float    f4;

#define B_  16
#define Q_  8
#define S_  4096
#define D_  2048
#define H_  16
#define F_  128
#define SQ_ 4104   // S + Q
#define HF_ 2048   // H * F
#define M_  128    // B * Q
#define NW_ 8      // waves per attention block (split-K over the key axis)
#define NCHUNK_ 129  // ceil(SQ_/32)

// Non-temporal helpers: all cache_k/cache_v traffic is single-use streaming;
// keep it out of the 192MB L2 so weights/x/q stay resident.
__device__ __forceinline__ f4 ldnt(const float* p) {
  return __builtin_nontemporal_load((const f4*)p);
}

// Wave-local LDS ordering (waves in the attention block run different trip
// counts, so __syncthreads() is not usable inside the chunk loop).
__device__ __forceinline__ void wave_lds_fence() {
  asm volatile("s_wait_dscnt 0" ::: "memory");
}

// Pack 16 contiguous floats (as 4 f4) into a v16h with optional scale.
__device__ __forceinline__ v16h pack16(f4 a0, f4 a1, f4 a2, f4 a3, float s) {
  v16h r;
  r[0]  = (_Float16)(a0[0] * s); r[1]  = (_Float16)(a0[1] * s);
  r[2]  = (_Float16)(a0[2] * s); r[3]  = (_Float16)(a0[3] * s);
  r[4]  = (_Float16)(a1[0] * s); r[5]  = (_Float16)(a1[1] * s);
  r[6]  = (_Float16)(a1[2] * s); r[7]  = (_Float16)(a1[3] * s);
  r[8]  = (_Float16)(a2[0] * s); r[9]  = (_Float16)(a2[1] * s);
  r[10] = (_Float16)(a2[2] * s); r[11] = (_Float16)(a2[3] * s);
  r[12] = (_Float16)(a3[0] * s); r[13] = (_Float16)(a3[1] * s);
  r[14] = (_Float16)(a3[2] * s); r[15] = (_Float16)(a3[3] * s);
  return r;
}

// ---------------------------------------------------------------------------
// Kernel 1: stream cache_k / cache_v into the (B, S+Q, H, F) output buffers.
// Pure bandwidth: f4 grid-stride copy, NT on both sides (single-use data).
// ---------------------------------------------------------------------------
__global__ __launch_bounds__(256)
void cdna5_cache_copy(const f4* __restrict__ src, f4* __restrict__ dst,
                      long long per_src, long long per_dst, long long total) {
  long long i = (long long)blockIdx.x * blockDim.x + threadIdx.x;
  const long long stride = (long long)gridDim.x * blockDim.x;
  for (; i < total; i += stride) {
    const long long b = i / per_src;
    const long long r = i - b * per_src;
    const f4 v = __builtin_nontemporal_load(src + i);
    __builtin_nontemporal_store(v, dst + b * per_dst + r);
  }
}

// ---------------------------------------------------------------------------
// Kernel 2: C[M_ x 2048] = A[M_ x 2048] * W[2048 x 2048] + bias, fp16 WMMA,
// fp32 accumulate. One wave per 16x16 output tile, K stepped by 32.
// cache_layout=1 scatters rows into the (B, S+Q, H, F) cache at rows S..S+Q-1.
// ---------------------------------------------------------------------------
__global__ __launch_bounds__(32)
void cdna5_wmma_gemm(const float* __restrict__ A, const float* __restrict__ W,
                     const float* __restrict__ bias, float* __restrict__ C,
                     int cache_layout) {
  __shared__ float lds_b[32][16];
  const int lane = threadIdx.x & 31;
  const int g    = lane >> 4;   // half-group (0/1)
  const int ln   = lane & 15;
  const int n0   = blockIdx.x * 16;
  const int m0   = blockIdx.y * 16;

  v8f acc = {};
  for (int k0 = 0; k0 < 2048; k0 += 32) {
    // Stage B tile (32 x 16) through LDS: lane loads one W row, coalesced.
    const float* wrow = W + (long long)(k0 + lane) * 2048 + n0;
    *(f4*)&lds_b[lane][0]  = *(const f4*)(wrow + 0);
    *(f4*)&lds_b[lane][4]  = *(const f4*)(wrow + 4);
    *(f4*)&lds_b[lane][8]  = *(const f4*)(wrow + 8);
    *(f4*)&lds_b[lane][12] = *(const f4*)(wrow + 12);
    __syncthreads();

    // A tile in wave32 A-layout: lane row = m0+ln, half e -> k0+16g+e:
    // one contiguous 16-float run per lane.
    const float* arow = A + (long long)(m0 + ln) * 2048 + k0 + 16 * g;
    v16h av = pack16(*(const f4*)(arow + 0),  *(const f4*)(arow + 4),
                     *(const f4*)(arow + 8), *(const f4*)(arow + 12), 1.0f);

    // B in wave32 B-layout: lane col = ln, half e -> k row 16*g + e.
    v16h bv;
#pragma unroll
    for (int e = 0; e < 16; ++e) bv[e] = (_Float16)lds_b[16 * g + e][ln];

    acc = __builtin_amdgcn_wmma_f32_16x16x32_f16(false, av, false, bv,
                                                 (short)0, acc, false, false);
    __syncthreads();
  }

  // Branch-once epilogue: within this tile the batch index is uniform.
  // C-layout rows are m0 + r + 8g, so row>>3 == m0/8 + g and row&7 == r.
  const int col = n0 + ln;
  const float bval = bias[col];
  float* crow;
  if (cache_layout) {
    crow = C + ((long long)((m0 >> 3) + g) * SQ_ + S_) * 2048 + col;
  } else {
    crow = C + (long long)(m0 + 8 * g) * 2048 + col;
  }
#pragma unroll
  for (int r = 0; r < 8; ++r) crow[(long long)r * 2048] = acc[r] + bval;
}

// ---------------------------------------------------------------------------
// Kernel 3: flash-decode attention, split-K over the key axis.
// One 256-thread block (8 waves) per (b, h). Wave w flash-processes key
// chunks w, w+8, w+16, ... (32 keys each) with private LDS slices, then the
// block merges the 8 partial softmax states (max/sum/accumulator) in LDS.
// 8x the memory-level parallelism of one-wave-per-(b,h): this phase moves
// ~1.05 GB of the ~2.1 GB total and sets the kernel's bandwidth floor.
// ---------------------------------------------------------------------------
__global__ __launch_bounds__(256)
void cdna5_flash_decode(const float* __restrict__ qws, const float* __restrict__ kcache,
                        const float* __restrict__ vcache, float* __restrict__ attn) {
  __shared__ _Float16 lds_p[NW_][16][32];    // probs: [wave][q_row][key_in_chunk]
  __shared__ _Float16 lds_v[NW_][32][128];   // V chunk per wave (fp16)
  __shared__ float    lds_o[NW_][8][128];    // partial O (real rows only)
  __shared__ float    lds_m[NW_][8];         // partial row max
  __shared__ float    lds_l[NW_][8];         // partial row sum
  const int lane = threadIdx.x & 31;
  const int wid  = threadIdx.x >> 5;         // 0..7
  const int g    = lane >> 4;
  const int ln   = lane & 15;
  const int b    = blockIdx.x >> 4;          // H_ == 16
  const int h    = blockIdx.x & 15;
  const float qscale = 0.088388347648318447f;  // 1/sqrt(F)

  // Preload q tile (16 rows, rows >= Q zero-padded) in A-layout, pre-scaled.
  v16h qa[4];
  {
    const int m = ln;
    const float* qrow = qws + (long long)(b * Q_ + (m & 7)) * HF_ + h * F_;
#pragma unroll
    for (int c = 0; c < 4; ++c) {
      f4 a0, a1, a2, a3;
      if (m < Q_) {
        const float* p = qrow + 32 * c + 16 * g;
        a0 = *(const f4*)(p + 0);
        a1 = *(const f4*)(p + 4);
        a2 = *(const f4*)(p + 8);
        a3 = *(const f4*)(p + 12);
      } else {
        a0 = a1 = a2 = a3 = f4{0.f, 0.f, 0.f, 0.f};
      }
      qa[c] = pack16(a0, a1, a2, a3, qscale);
    }
  }

  float Mrun[8], Lrun[8];
  v8f o[8];
  {
    v8f z = {};
#pragma unroll
    for (int r = 0; r < 8; ++r) { Mrun[r] = -3.0e38f; Lrun[r] = 0.f; }
#pragma unroll
    for (int t = 0; t < 8; ++t) o[t] = z;
  }

  const long long bh = (long long)b * SQ_ * HF_ + h * F_;

  for (int ci = wid; ci < NCHUNK_; ci += NW_) {
    const int base = ci * 32;
    v8f s0 = {}, s1 = {};
    const int key0 = base + ln;
    const int key1 = base + 16 + ln;
    const int k0c  = key0 < SQ_ ? key0 : SQ_ - 1;
    const int k1c  = key1 < SQ_ ? key1 : SQ_ - 1;

    // scores = q * K^T : B[f][key]; lane col = key, half e -> f = 32c+16g+e.
    // K rows are single-use: non-temporal loads.
#pragma unroll
    for (int c = 0; c < 4; ++c) {
      {
        const float* p = kcache + bh + (long long)k0c * HF_ + 32 * c + 16 * g;
        v16h kb = pack16(ldnt(p + 0), ldnt(p + 4), ldnt(p + 8), ldnt(p + 12), 1.0f);
        s0 = __builtin_amdgcn_wmma_f32_16x16x32_f16(false, qa[c], false, kb,
                                                    (short)0, s0, false, false);
      }
      {
        const float* p = kcache + bh + (long long)k1c * HF_ + 32 * c + 16 * g;
        v16h kb = pack16(ldnt(p + 0), ldnt(p + 4), ldnt(p + 8), ldnt(p + 12), 1.0f);
        s1 = __builtin_amdgcn_wmma_f32_16x16x32_f16(false, qa[c], false, kb,
                                                    (short)0, s1, false, false);
      }
    }
    if (key0 >= SQ_) {
#pragma unroll
      for (int r = 0; r < 8; ++r) s0[r] = -3.0e38f;
    }
    if (key1 >= SQ_) {
#pragma unroll
      for (int r = 0; r < 8; ++r) s1[r] = -3.0e38f;
    }

    // Online softmax: row m = r + 8g spans the 16 lanes of a half-group.
#pragma unroll
    for (int r = 0; r < 8; ++r) {
      float mx = fmaxf(s0[r], s1[r]);
      mx = fmaxf(mx, __shfl_xor(mx, 1, 16));
      mx = fmaxf(mx, __shfl_xor(mx, 2, 16));
      mx = fmaxf(mx, __shfl_xor(mx, 4, 16));
      mx = fmaxf(mx, __shfl_xor(mx, 8, 16));
      const float Mn = fmaxf(Mrun[r], mx);
      const float sc = __expf(Mrun[r] - Mn);
      const float p0 = __expf(s0[r] - Mn);
      const float p1 = __expf(s1[r] - Mn);
      float rs = p0 + p1;
      rs += __shfl_xor(rs, 1, 16);
      rs += __shfl_xor(rs, 2, 16);
      rs += __shfl_xor(rs, 4, 16);
      rs += __shfl_xor(rs, 8, 16);
      Lrun[r] = Lrun[r] * sc + rs;
      Mrun[r] = Mn;
      s0[r] = p0;
      s1[r] = p1;
#pragma unroll
      for (int t = 0; t < 8; ++t) o[t][r] *= sc;
    }

    // C-layout -> A-layout relayout of P through this wave's LDS slice;
    // stage V chunk (non-temporal read) as fp16.
#pragma unroll
    for (int r = 0; r < 8; ++r) {
      lds_p[wid][r + 8 * g][ln]      = (_Float16)s0[r];
      lds_p[wid][r + 8 * g][16 + ln] = (_Float16)s1[r];
    }
    const int vkey = (base + lane) < SQ_ ? (base + lane) : SQ_ - 1;
    const float* vrow = vcache + bh + (long long)vkey * HF_;
#pragma unroll
    for (int f8 = 0; f8 < 32; ++f8) {
      const f4 vv = ldnt(vrow + 4 * f8);
      v4h hv;
      hv[0] = (_Float16)vv[0]; hv[1] = (_Float16)vv[1];
      hv[2] = (_Float16)vv[2]; hv[3] = (_Float16)vv[3];
      *(v4h*)&lds_v[wid][lane][4 * f8] = hv;
    }
    wave_lds_fence();  // per-wave LDS slice: wave-local ordering only

    v16h pa;
#pragma unroll
    for (int e = 0; e < 16; ++e) {
      const int kk = 16 * g + e;  // A-layout K index
      pa[e] = lds_p[wid][ln][kk];
    }
#pragma unroll
    for (int t = 0; t < 8; ++t) {
      v16h vb;
#pragma unroll
      for (int e = 0; e < 16; ++e) vb[e] = lds_v[wid][16 * g + e][16 * t + ln];
      o[t] = __builtin_amdgcn_wmma_f32_16x16x32_f16(false, pa, false, vb,
                                                    (short)0, o[t], false, false);
    }
    wave_lds_fence();  // reads done before next iteration overwrites slices
  }

  // Publish this wave's partial state (un-normalized O, row max, row sum).
  if (g == 0) {
#pragma unroll
    for (int r = 0; r < 8; ++r) {
#pragma unroll
      for (int t = 0; t < 8; ++t) lds_o[wid][r][16 * t + ln] = o[t][r];
    }
    if (ln == 0) {
#pragma unroll
      for (int r = 0; r < 8; ++r) { lds_m[wid][r] = Mrun[r]; lds_l[wid][r] = Lrun[r]; }
    }
  }
  __syncthreads();

  // Split-softmax merge across the 8 waves: global max, rescale, normalize.
  for (int idx = threadIdx.x; idx < Q_ * F_; idx += 256) {
    const int m = idx >> 7;          // q row
    const int f = idx & 127;         // feature
    float gm = -3.0e38f;
#pragma unroll
    for (int w = 0; w < NW_; ++w) gm = fmaxf(gm, lds_m[w][m]);
    float gl = 0.f, acc = 0.f;
#pragma unroll
    for (int w = 0; w < NW_; ++w) {
      const float sc = __expf(lds_m[w][m] - gm);
      gl  += lds_l[w][m] * sc;
      acc += lds_o[w][m][f] * sc;
    }
    attn[(long long)(b * Q_ + m) * HF_ + h * F_ + f] = acc / gl;
  }
}

// ---------------------------------------------------------------------------
extern "C" void kernel_launch(void* const* d_in, const int* in_sizes, int n_in,
                              void* d_out, int out_size, void* d_ws, size_t ws_size,
                              hipStream_t stream) {
  (void)in_sizes; (void)n_in; (void)out_size; (void)ws_size;

  const float* x  = (const float*)d_in[0];
  const float* ck = (const float*)d_in[1];
  const float* cv = (const float*)d_in[2];
  const float* wq = (const float*)d_in[3];
  const float* bq = (const float*)d_in[4];
  const float* wk = (const float*)d_in[5];
  const float* bk = (const float*)d_in[6];
  const float* wv = (const float*)d_in[7];
  const float* bv = (const float*)d_in[8];
  const float* wo = (const float*)d_in[9];
  const float* bo = (const float*)d_in[10];

  float* out  = (float*)d_out;                       // [B,Q,D]
  float* kout = out + (long long)M_ * D_;            // [B,S+Q,H,F]
  float* vout = kout + (long long)B_ * SQ_ * HF_;    // [B,S+Q,H,F]
  float* qws  = (float*)d_ws;                        // q   [M_, HF_]  (1 MB)
  float* aws  = qws + (long long)M_ * HF_;           // attn[M_, HF_]  (1 MB)

  // 1) Copy old cache rows into the output k/v buffers (bandwidth-bound part).
  const long long per_src = (long long)S_ * HF_ / 4;
  const long long per_dst = (long long)SQ_ * HF_ / 4;
  const long long tot4    = (long long)B_ * per_src;
  cdna5_cache_copy<<<32768, 256, 0, stream>>>((const f4*)ck, (f4*)kout,
                                              per_src, per_dst, tot4);
  cdna5_cache_copy<<<32768, 256, 0, stream>>>((const f4*)cv, (f4*)vout,
                                              per_src, per_dst, tot4);

  // 2) QKV projections (WMMA). k_new/v_new land directly in the cache tails.
  dim3 gg(HF_ / 16, M_ / 16);  // (128, 8) tiles
  cdna5_wmma_gemm<<<gg, 32, 0, stream>>>(x, wq, bq, qws, 0);
  cdna5_wmma_gemm<<<gg, 32, 0, stream>>>(x, wk, bk, kout, 1);
  cdna5_wmma_gemm<<<gg, 32, 0, stream>>>(x, wv, bv, vout, 1);

  // 3) Flash-decode attention, 8-way split-K per (b,h): 2048 streaming waves.
  cdna5_flash_decode<<<B_ * H_, 256, 0, stream>>>(qws, kout, vout, aws);

  // 4) Output projection (WMMA): attn [M_,HF_] x wo [HF_,D_] + bo -> out.
  cdna5_wmma_gemm<<<gg, 32, 0, stream>>>(aws, wo, bo, out, 0);
}
